// ScaledDotProductAttention_15049565405453
// MI455X (gfx1250) — matrix operand order repaired and checked by reading, not compiled, not run on hardware
//
#include <hip/hip_runtime.h>

// FlashAttention-style scaled dot-product attention for gfx1250 (MI455X).
// B=16, N=2048, D=64, fp32 in/out, f16 WMMA with f32 accumulation.
// Round 5: native v_exp_f32 via __builtin_amdgcn_exp2f (drops ldexp fixups).

typedef __attribute__((ext_vector_type(16))) _Float16 f16x16;
typedef __attribute__((ext_vector_type(8)))  _Float16 f16x8;
typedef __attribute__((ext_vector_type(8)))  float    f32x8;

constexpr int AB  = 16;    // batch
constexpr int AN  = 2048;  // sequence length
constexpr int AD  = 64;    // head dim
constexpr int KBLK = 32;   // keys per iteration
constexpr int WAVES = 4;   // waves per workgroup (16 query rows each)

__device__ __forceinline__ f16x16 cat8(f16x8 lo, f16x8 hi) {
  f16x16 r;
#pragma unroll
  for (int i = 0; i < 8; ++i) { r[i] = lo[i]; r[i + 8] = hi[i]; }
  return r;
}

// Load 8 contiguous f32, scale, convert to f16x8.
__device__ __forceinline__ f16x8 cvt8(const float* __restrict__ p, float s) {
  const float4 a = *(const float4*)p;
  const float4 b = *(const float4*)(p + 4);
  f16x8 r;
  r[0] = (_Float16)(a.x * s); r[1] = (_Float16)(a.y * s);
  r[2] = (_Float16)(a.z * s); r[3] = (_Float16)(a.w * s);
  r[4] = (_Float16)(b.x * s); r[5] = (_Float16)(b.y * s);
  r[6] = (_Float16)(b.z * s); r[7] = (_Float16)(b.w * s);
  return r;
}

__device__ __forceinline__ f32x8 wmma_f16(f16x16 a, f16x16 b, f32x8 c) {
  // D = A(16x32 f16) * B(32x16 f16) + C(16x16 f32)
  return __builtin_amdgcn_wmma_f32_16x16x32_f16(
      false, a, false, b, (short)0, c, false, false);
}

// Native v_exp_f32 (2^x). Exact for our range: args are in [-inf, 0] and
// weights below 2^-30 are numerically irrelevant; exp2(-huge) -> 0.
__device__ __forceinline__ float fast_exp2(float x) {
  return __builtin_amdgcn_exp2f(x);
}

// Plain v_max without LLVM's sNaN canonicalization overhead.
__device__ __forceinline__ float vmaxa(float a, float b) {
  float d;
  asm("v_max_num_f32_e32 %0, %1, %2" : "=v"(d) : "v"(a), "v"(b));
  return d;
}

// Row-max over each 16-lane group, one fused DPP-max per stage:
// combine lanes i^1, i^2, 7-i, 15-i  ->  full 16-lane max in 4 VALU ops.
__device__ __forceinline__ float rowmax16(float x) {
  asm("v_max_num_f32_dpp %0, %0, %0 quad_perm:[1,0,3,2] row_mask:0xf bank_mask:0xf"
      : "+v"(x));
  asm("v_max_num_f32_dpp %0, %0, %0 quad_perm:[2,3,0,1] row_mask:0xf bank_mask:0xf"
      : "+v"(x));
  asm("v_max_num_f32_dpp %0, %0, %0 row_half_mirror row_mask:0xf bank_mask:0xf"
      : "+v"(x));
  asm("v_max_num_f32_dpp %0, %0, %0 row_mirror row_mask:0xf bank_mask:0xf"
      : "+v"(x));
  return x;
}

// Per-thread register staging of one 32x64 K/V tile slice (f16).
struct Stage {
  f16x8 k0, k1;
  _Float16 vh[16];
};

__device__ __forceinline__ Stage load_stage(const float* __restrict__ kg,
                                            const float* __restrict__ vg,
                                            size_t rbase) {
  Stage st;
  st.k0 = cvt8(kg + rbase,     1.0f);
  st.k1 = cvt8(kg + rbase + 8, 1.0f);
  const float* vp = vg + rbase;
  const float4 va = *(const float4*)(vp);
  const float4 vb = *(const float4*)(vp + 4);
  const float4 vc = *(const float4*)(vp + 8);
  const float4 vd = *(const float4*)(vp + 12);
  const float vv[16] = {va.x, va.y, va.z, va.w, vb.x, vb.y, vb.z, vb.w,
                        vc.x, vc.y, vc.z, vc.w, vd.x, vd.y, vd.z, vd.w};
#pragma unroll
  for (int j = 0; j < 16; ++j) st.vh[j] = (_Float16)vv[j];
  return st;
}

__global__ __launch_bounds__(128)
void fa_fwd_kernel(const float* __restrict__ qg,
                   const float* __restrict__ kg,
                   const float* __restrict__ vg,
                   float* __restrict__ og) {
  // LDS staging. Row padding keeps the 16-lane strided reads on 16 banks.
  __shared__ _Float16 sK [KBLK][AD + 8];         // [key][d]   row-major f16
  __shared__ _Float16 sVt[AD][KBLK + 8];         // [d][perm-key] transposed f16
  __shared__ unsigned int sP2[WAVES][16][20];    // packed P: {tile0,tile1} f16 pairs

  const int lane = threadIdx.x & 31;
  const int w    = threadIdx.x >> 5;
  const int bi   = blockIdx.x >> 5;   // batch index   (32 blocks per batch)
  const int tile = blockIdx.x & 31;   // 64-row tile within batch
  const int qbase = tile * 64 + w * 16;

  const int m16   = lane & 15;        // column / row-in-tile index
  const int hi    = lane >> 4;        // half-wave select
  const int off8  = hi * 8;           // C-layout row base (accumulator rows)
  const int off16 = hi * 16;          // B-fragment K-chunk base

  // ---- Q fragments (A-matrix 16x32 f16) ----
  // Fold 1/sqrt(D) AND log2(e) into Q: softmax computed in exp2 domain,
  // exp2(s*log2e) == exp(s) exactly, so results are identical.
  const float scale = 0.125f * 1.44269504088896340736f;
  const float* qrow = qg + (size_t)(bi * AN + qbase + m16) * AD;
  const f16x16 aQ0 = cat8(cvt8(qrow + off16,      scale),
                          cvt8(qrow + 8 + off16, scale));
  const f16x16 aQ1 = cat8(cvt8(qrow + 32 + off16, scale),
                          cvt8(qrow + 40 + off16, scale));

  // Ones B-matrix: P x ones accumulates the softmax denominator as a 5th
  // O accumulator (every column of the product equals the row sum of P).
  f16x16 bOnes;
#pragma unroll
  for (int i = 0; i < 16; ++i) bOnes[i] = (_Float16)1.0f;

  f32x8 o0 = {}, o1 = {}, o2 = {}, o3 = {}, o4 = {};
  float m_run[8];
#pragma unroll
  for (int r = 0; r < 8; ++r) m_run[r] = -1e30f;

  // cooperative staging indices: 128 threads cover 32 keys x 64 dims
  const int kr = threadIdx.x >> 2;        // key row 0..31
  const int dc = (threadIdx.x & 3) << 4;  // d column base 0,16,32,48
  // Packed-P readback induces key order key(k)=8(k>>4)+((k&15)>>1)+16(k&1) on
  // the PV matmul; stage V row t into matrix row kinv(t) to compensate.
  const int kvcol = 16 * ((kr >> 3) & 1) + 2 * (kr & 7) + (kr >> 4);
  const size_t kvstride = (size_t)(bi * AN + kr) * AD + dc;

  Stage st = load_stage(kg, vg, kvstride);  // prefetch tile 0

  for (int kb = 0; kb < AN; kb += KBLK) {
    // ---- commit prefetched tile to LDS: K row-major, V transposed+permuted ----
    *(f16x8*)&sK[kr][dc]     = st.k0;
    *(f16x8*)&sK[kr][dc + 8] = st.k1;
#pragma unroll
    for (int j = 0; j < 16; ++j)
      sVt[dc + j][kvcol] = st.vh[j];
    __syncthreads();

    // ---- prefetch next tile while this one is consumed ----
    if (kb + KBLK < AN)
      st = load_stage(kg, vg, kvstride + (size_t)(kb + KBLK) * AD);

    // ---- S = Q K^T : two 16x16 f32 tiles covering 32 keys ----
    // B-fragment (32x16): lane n holds K = off16..off16+15 contiguous.
    const _Float16* kb0 = &sK[m16][off16];        // keys kb..kb+15
    const _Float16* kb1 = &sK[16 + m16][off16];   // keys kb+16..kb+31
    const f16x16 bK00 = cat8(*(const f16x8*)(kb0),      *(const f16x8*)(kb0 + 8));
    const f16x16 bK01 = cat8(*(const f16x8*)(kb0 + 32), *(const f16x8*)(kb0 + 40));
    const f16x16 bK10 = cat8(*(const f16x8*)(kb1),      *(const f16x8*)(kb1 + 8));
    const f16x16 bK11 = cat8(*(const f16x8*)(kb1 + 32), *(const f16x8*)(kb1 + 40));

    f32x8 s0 = {}, s1 = {};
    s0 = wmma_f16(aQ0, bK00, s0);
    s0 = wmma_f16(aQ1, bK01, s0);
    s1 = wmma_f16(aQ0, bK10, s1);
    s1 = wmma_f16(aQ1, bK11, s1);

    // ---- online softmax over this 32-key block (exp2 domain) ----
    // C-layout: element r of s0/s1 is row (qbase + off8 + r), col (kb + m16 [+16]).
    float bm[8];
#pragma unroll
    for (int r = 0; r < 8; ++r) bm[r] = rowmax16(vmaxa(s0[r], s1[r]));

    bool grow = false;
#pragma unroll
    for (int r = 0; r < 8; ++r) grow = grow || (bm[r] > m_run[r]);
    if (__any(grow)) {  // rare after warm-up: rescale only when max advances
#pragma unroll
      for (int r = 0; r < 8; ++r) {
        const float nm = vmaxa(m_run[r], bm[r]);
        const float sc = fast_exp2(m_run[r] - nm);
        m_run[r] = nm;
        o0[r] *= sc; o1[r] *= sc; o2[r] *= sc; o3[r] *= sc; o4[r] *= sc;
      }
    }

    float p0v[8], p1v[8];
#pragma unroll
    for (int r = 0; r < 8; ++r) {
      p0v[r] = fast_exp2(s0[r] - m_run[r]);
      p1v[r] = fast_exp2(s1[r] - m_run[r]);
    }

    // ---- packed P transpose (C-layout -> A-layout) through per-wave LDS ----
#pragma unroll
    for (int r = 0; r < 8; ++r) {
      union { _Float16 h[2]; unsigned int u; } pk;
      pk.h[0] = (_Float16)p0v[r];   // tile0 col m16      (v_cvt_pk_f16_f32)
      pk.h[1] = (_Float16)p1v[r];   // tile1 col m16+16
      sP2[w][off8 + r][m16] = pk.u;
    }
    // Same-wave LDS ops complete in order; wait + compiler fence before reload.
    asm volatile("s_wait_dscnt 0" ::: "memory");
    const _Float16* pp = (const _Float16*)&sP2[w][m16][0];
    const f16x16 aP = cat8(*(const f16x8*)(pp + off16),
                           *(const f16x8*)(pp + 8 + off16));

    // ---- O += P * [V | ones] : four 16-wide D chunks + denominator ----
    const _Float16* v0p = &sVt[m16][off16];
    const _Float16* v1p = &sVt[16 + m16][off16];
    const _Float16* v2p = &sVt[32 + m16][off16];
    const _Float16* v3p = &sVt[48 + m16][off16];
    const f16x16 bV0 = cat8(*(const f16x8*)v0p, *(const f16x8*)(v0p + 8));
    const f16x16 bV1 = cat8(*(const f16x8*)v1p, *(const f16x8*)(v1p + 8));
    const f16x16 bV2 = cat8(*(const f16x8*)v2p, *(const f16x8*)(v2p + 8));
    const f16x16 bV3 = cat8(*(const f16x8*)v3p, *(const f16x8*)(v3p + 8));
    o0 = wmma_f16(aP, bV0, o0);
    o1 = wmma_f16(aP, bV1, o1);
    o2 = wmma_f16(aP, bV2, o2);
    o3 = wmma_f16(aP, bV3, o3);
    o4 = wmma_f16(aP, bOnes, o4);   // row sums of P

    __syncthreads();  // protect sK/sVt before next overwrite
  }

  // ---- normalize and store (C-layout scatter) ----
#pragma unroll
  for (int r = 0; r < 8; ++r) {
    const float inv = 1.0f / o4[r];
    const size_t row = (size_t)(bi * AN + qbase + off8 + r) * AD;
    og[row + m16]      = o0[r] * inv;
    og[row + 16 + m16] = o1[r] * inv;
    og[row + 32 + m16] = o2[r] * inv;
    og[row + 48 + m16] = o3[r] * inv;
  }
}

extern "C" void kernel_launch(void* const* d_in, const int* in_sizes, int n_in,
                              void* d_out, int out_size, void* d_ws, size_t ws_size,
                              hipStream_t stream) {
  (void)in_sizes; (void)n_in; (void)out_size; (void)d_ws; (void)ws_size;
  const float* q = (const float*)d_in[0];
  const float* k = (const float*)d_in[1];
  const float* v = (const float*)d_in[2];
  float* out = (float*)d_out;

  const dim3 grid(AB * (AN / (16 * WAVES)));  // 16 * 32 = 512 blocks
  const dim3 block(32 * WAVES);               // 128 threads = 4 waves
  fa_fwd_kernel<<<grid, block, 0, stream>>>(q, k, v, out);
}
